// Encoder_Core_78563541778978
// MI455X (gfx1250) — matrix operand order repaired and checked by reading, not compiled, not run on hardware
//
#include <hip/hip_runtime.h>
#include <hip/hip_bf16.h>

#define N_NODES 100000
#define N_EDGES 1600000
#define DIM     128
#define NLAYERS 3
#define NGRAPH  512
#define PDIM    384   // L * D

typedef __attribute__((ext_vector_type(16))) __bf16         v16bf;
typedef __attribute__((ext_vector_type(8)))  float          v8f;
typedef __attribute__((ext_vector_type(8)))  unsigned short v8us;
typedef __attribute__((ext_vector_type(16))) unsigned short v16us;

__device__ __forceinline__ unsigned short f2bf(float f) {
  unsigned int u = __float_as_uint(f);
  u += 0x7fffu + ((u >> 16) & 1u);          // round-to-nearest-even
  return (unsigned short)(u >> 16);
}

// A operand: two contiguous 16B chunks (k..k+7 | k+16..k+23) -> v16bf
__device__ __forceinline__ v16bf loadA(const unsigned short* lo,
                                       const unsigned short* hi) {
  union { v16us w; v8us h[2]; } u;
  u.h[0] = *(const v8us*)lo;
  u.h[1] = *(const v8us*)hi;
  return __builtin_bit_cast(v16bf, u.w);
}
// B operand: one contiguous 32B chunk (pre-swizzled) -> v16bf
__device__ __forceinline__ v16bf loadB(const unsigned short* p) {
  return __builtin_bit_cast(v16bf, *(const v16us*)p);
}

// ---------------------------------------------------------------- utilities
__global__ void zero_f32_kernel(float* __restrict__ p, int n) {
  for (int i = blockIdx.x * blockDim.x + threadIdx.x; i < n;
       i += gridDim.x * blockDim.x)
    p[i] = 0.0f;
}

__global__ void cvt_bf16_kernel(const float* __restrict__ s,
                                unsigned short* __restrict__ d, int n) {
  int i = blockIdx.x * blockDim.x + threadIdx.x;
  if (i < n) d[i] = f2bf(s[i]);
}

// Pre-swizzle a row-major f32 weight [K][N] into WMMA-B lane layout:
// out[((nt*(K/32)+kt)*32 + lane)*16 + i] = bf16(W[kt*32+(lane>>4)*8+(i&7)+(i>>3)*16][nt*16+(lane&15)])
// -> each lane's 16 B elements are 32 contiguous bytes (2x global_load_b128).
__global__ void swizzleB_kernel(const float* __restrict__ W,
                                unsigned short* __restrict__ out, int K, int N) {
  int idx = blockIdx.x * blockDim.x + threadIdx.x;
  if (idx >= K * N) return;
  int i    = idx & 15;
  int lane = (idx >> 4) & 31;
  int t    = idx >> 9;                 // nt*(K/32) + kt
  int ktiles = K >> 5;
  int nt = t / ktiles;
  int kt = t - nt * ktiles;
  int k  = kt * 32 + (lane >> 4) * 8 + (i & 7) + (i >> 3) * 16;
  int n  = nt * 16 + (lane & 15);
  out[idx] = f2bf(W[(size_t)k * N + n]);
}

// ----------------------------------------------------- edge scatter (GIN agg)
__global__ __launch_bounds__(256) void scatter_add_kernel(
    const float* __restrict__ h, const int* __restrict__ src,
    const int* __restrict__ dst, float* __restrict__ agg) {
  int tid = blockIdx.x * 256 + threadIdx.x;
  if (tid >= N_EDGES * 32) return;
  int e = tid >> 5;
  int f = (tid & 31) << 2;
  int s = src[e];
  int d = dst[e];
  float4 v = *(const float4*)(h + (size_t)s * DIM + f);
  float* p = agg + (size_t)d * DIM + f;
  atomicAdd(p + 0, v.x);
  atomicAdd(p + 1, v.y);
  atomicAdd(p + 2, v.z);
  atomicAdd(p + 3, v.w);
}

// --------------------------------------------------- fused GIN MLP via WMMA
// Block: 16 rows x 128 cols, 8 waves (one 16x16 tile per wave).
// z = relu( relu((h+agg) @ W1 + b1) @ W2 + b2 ), fused through LDS.
__global__ __launch_bounds__(256) void gin_mlp_kernel(
    const float* __restrict__ h, const float* __restrict__ agg,
    const unsigned short* __restrict__ w1s, const float* __restrict__ b1,
    const unsigned short* __restrict__ w2s, const float* __restrict__ b2,
    float* __restrict__ z) {
  __shared__ unsigned short X[16][DIM + 8];   // row stride 272B (16B aligned)
  __shared__ unsigned short Y[16][DIM + 8];
  const int r0  = blockIdx.x * 16;
  const int tid = threadIdx.x;

  // stage bf16(h + agg): float4 reads, packed uint2 LDS stores
  for (int i = tid; i < 16 * (DIM / 4); i += 256) {
    int r  = i >> 5;                 // DIM/4 = 32 quads per row
    int c4 = (i & 31) << 2;
    size_t g = (size_t)(r0 + r) * DIM + c4;
    float4 hv = *(const float4*)(h + g);
    float4 av = *(const float4*)(agg + g);
    unsigned int lo = (unsigned int)f2bf(hv.x + av.x) |
                      ((unsigned int)f2bf(hv.y + av.y) << 16);
    unsigned int hi = (unsigned int)f2bf(hv.z + av.z) |
                      ((unsigned int)f2bf(hv.w + av.w) << 16);
    *(uint2*)&X[r][c4] = make_uint2(lo, hi);
  }
  __syncthreads();

  const int lane = tid & 31;
  const int wave = tid >> 5;        // n-tile (16-col window)
  const int m    = lane & 15;
  const int hk   = lane >> 4;
  const int col  = wave * 16 + m;

  // ---- GEMM 1
  v8f acc = {};
#pragma unroll
  for (int kt = 0; kt < 4; ++kt) {
    v16bf a = loadA(&X[m][kt * 32 + hk * 8], &X[m][kt * 32 + 16 + hk * 8]);
    v16bf b = loadB(w1s + (((wave * 4 + kt) * 32 + lane) << 4));
    acc = __builtin_amdgcn_wmma_f32_16x16x32_bf16(false, a, false, b,
                                                  (short)0, acc, false, false);
  }
  const float bias1 = b1[col];
#pragma unroll
  for (int v = 0; v < 8; ++v)                       // D: M = v + hk*8
    Y[v + hk * 8][col] = f2bf(fmaxf(acc[v] + bias1, 0.0f));
  __syncthreads();

  // ---- GEMM 2
  v8f acc2 = {};
#pragma unroll
  for (int kt = 0; kt < 4; ++kt) {
    v16bf a = loadA(&Y[m][kt * 32 + hk * 8], &Y[m][kt * 32 + 16 + hk * 8]);
    v16bf b = loadB(w2s + (((wave * 4 + kt) * 32 + lane) << 4));
    acc2 = __builtin_amdgcn_wmma_f32_16x16x32_bf16(false, a, false, b,
                                                   (short)0, acc2, false, false);
  }
  const float bias2 = b2[col];
#pragma unroll
  for (int v = 0; v < 8; ++v) {
    float t = fmaxf(acc2[v] + bias2, 0.0f);         // outer ReLU
    z[(size_t)(r0 + v + hk * 8) * DIM + col] = t;
  }
}

// ----------------------------------------------- BatchNorm stats (sum, sumsq)
__global__ __launch_bounds__(256) void col_stats_kernel(
    const float* __restrict__ z, float* __restrict__ s,
    float* __restrict__ ss, int nrows) {
  __shared__ float ls[256], lss[256];
  int c    = threadIdx.x & (DIM - 1);
  int half = threadIdx.x >> 7;
  int rows_per_blk = (nrows + gridDim.x - 1) / gridDim.x;
  int rbeg = blockIdx.x * rows_per_blk;
  int rend = min(nrows, rbeg + rows_per_blk);
  float sum = 0.0f, sumsq = 0.0f;
  for (int r = rbeg + half; r < rend; r += 2) {
    float v = z[(size_t)r * DIM + c];
    sum += v;
    sumsq += v * v;
  }
  ls[threadIdx.x]  = sum;
  lss[threadIdx.x] = sumsq;
  __syncthreads();
  if (half == 0) {
    atomicAdd(&s[c],  ls[c]  + ls[c + DIM]);
    atomicAdd(&ss[c], lss[c] + lss[c + DIM]);
  }
}

// ------------------------------------- BN normalize (in-place) + add-pool
__global__ __launch_bounds__(256) void bn_apply_pool_kernel(
    const float* __restrict__ z, const float* __restrict__ s,
    const float* __restrict__ ss, const float* __restrict__ gamma,
    const float* __restrict__ beta, const int* __restrict__ batch,
    float* __restrict__ hout, float* __restrict__ pooled, int layer) {
  int idx = blockIdx.x * 256 + threadIdx.x;
  if (idx >= N_NODES * DIM) return;
  int r = idx >> 7, c = idx & (DIM - 1);
  const float invN = 1.0f / (float)N_NODES;
  float mu  = s[c] * invN;
  float var = ss[c] * invN - mu * mu;               // biased variance
  float v   = (z[idx] - mu) * __frsqrt_rn(var + 1e-5f) * gamma[c] + beta[c];
  hout[idx] = v;
  atomicAdd(&pooled[(size_t)batch[r] * PDIM + layer * DIM + c], v);
}

// ------------------------------------------- generic bf16 WMMA GEMM (head)
// A:[M,K] bf16 row-major, Bsw: pre-swizzled bf16, out [M,Nc].
template <bool RELU, bool OUTBF>
__global__ __launch_bounds__(256) void gemm_bf16_kernel(
    const unsigned short* __restrict__ A,
    const unsigned short* __restrict__ Bsw, const float* __restrict__ bias,
    float* __restrict__ Cf, unsigned short* __restrict__ Cb, int M, int Nc,
    int K) {
  __shared__ unsigned short As[16][40];             // row stride 80B
  const int r0   = blockIdx.x * 16;
  const int tid  = threadIdx.x;
  const int lane = tid & 31;
  const int wave = tid >> 5;
  const int m    = lane & 15;
  const int hk   = lane >> 4;
  const int nt   = blockIdx.y * 8 + wave;           // global n-tile
  const int col  = nt * 16 + m;
  const int ktiles = K >> 5;

  // staging indices: one dword (2 bf16) per thread
  const int sr = tid >> 4;
  const int sc = (tid & 15) << 1;

  v8f acc = {};
  for (int kt = 0; kt < ktiles; ++kt) {
    *(unsigned int*)&As[sr][sc] =
        *(const unsigned int*)&A[(size_t)(r0 + sr) * K + kt * 32 + sc];
    __syncthreads();
    v16bf a = loadA(&As[m][hk * 8], &As[m][16 + hk * 8]);
    v16bf b = loadB(Bsw + (((nt * ktiles + kt) * 32 + lane) << 4));
    acc = __builtin_amdgcn_wmma_f32_16x16x32_bf16(false, a, false, b,
                                                  (short)0, acc, false, false);
    __syncthreads();
  }
  const float bb = bias[col];
#pragma unroll
  for (int v = 0; v < 8; ++v) {
    int r = r0 + v + hk * 8;
    float t = acc[v] + bb;
    if (RELU) t = fmaxf(t, 0.0f);
    if (OUTBF) Cb[(size_t)r * Nc + col] = f2bf(t);
    else       Cf[(size_t)r * Nc + col] = t;
  }
}

// --------------------------------------------------------- row L2 normalize
__global__ __launch_bounds__(128) void l2norm_kernel(
    const float* __restrict__ y, const float* __restrict__ xcat,
    float* __restrict__ out) {
  __shared__ float red[4];
  int row = blockIdx.x;
  const float* src;
  float* dst;
  if (row < NGRAPH) {
    src = y + (size_t)row * PDIM;
    dst = out + (size_t)row * PDIM;
  } else {
    src = xcat + (size_t)(row - NGRAPH) * PDIM;
    dst = out + (size_t)NGRAPH * PDIM + (size_t)(row - NGRAPH) * PDIM;
  }
  float s = 0.0f;
  for (int i = threadIdx.x; i < PDIM; i += 128) {
    float v = src[i];
    s += v * v;
  }
#pragma unroll
  for (int o = 16; o > 0; o >>= 1) s += __shfl_down(s, o, 32);
  if ((threadIdx.x & 31) == 0) red[threadIdx.x >> 5] = s;
  __syncthreads();
  if (threadIdx.x == 0) {
    float t = red[0] + red[1] + red[2] + red[3];
    red[0] = fmaxf(sqrtf(t), 1e-12f);
  }
  __syncthreads();
  float inv = 1.0f / red[0];
  for (int i = threadIdx.x; i < PDIM; i += 128) dst[i] = src[i] * inv;
}

// ---------------------------------------------------------------- launcher
extern "C" void kernel_launch(void* const* d_in, const int* in_sizes, int n_in,
                              void* d_out, int out_size, void* d_ws,
                              size_t ws_size, hipStream_t stream) {
  const float* x     = (const float*)d_in[0];
  const int*   ei    = (const int*)d_in[1];
  const int*   batch = (const int*)d_in[2];
  const float* w1    = (const float*)d_in[3];
  const float* b1    = (const float*)d_in[4];
  const float* w2    = (const float*)d_in[5];
  const float* b2    = (const float*)d_in[6];
  const float* gamma = (const float*)d_in[7];
  const float* beta  = (const float*)d_in[8];
  const float* pw1   = (const float*)d_in[9];
  const float* pb1   = (const float*)d_in[10];
  const float* pw2   = (const float*)d_in[11];
  const float* pb2   = (const float*)d_in[12];
  const int* srcI = ei;
  const int* dstI = ei + N_EDGES;
  float* out = (float*)d_out;

  char* base = (char*)d_ws;
  size_t off = 0;
  auto alloc = [&](size_t bytes) -> void* {
    void* p = (void*)(base + off);
    off = (off + bytes + 255) & ~(size_t)255;
    return p;
  };
  float* agg    = (float*)alloc((size_t)N_NODES * DIM * 4);
  float* bufA   = (float*)alloc((size_t)N_NODES * DIM * 4);
  float* bufB   = (float*)alloc((size_t)N_NODES * DIM * 4);
  float* pooled = (float*)alloc((size_t)NGRAPH * PDIM * 4);
  float* yfin   = (float*)alloc((size_t)NGRAPH * PDIM * 4);
  unsigned short* xcat_bf = (unsigned short*)alloc((size_t)NGRAPH * PDIM * 2);
  unsigned short* y1bf    = (unsigned short*)alloc((size_t)NGRAPH * PDIM * 2);
  unsigned short* w1bf    = (unsigned short*)alloc((size_t)NLAYERS * DIM * DIM * 2);
  unsigned short* w2bf    = (unsigned short*)alloc((size_t)NLAYERS * DIM * DIM * 2);
  unsigned short* pw1bf   = (unsigned short*)alloc((size_t)PDIM * PDIM * 2);
  unsigned short* pw2bf   = (unsigned short*)alloc((size_t)PDIM * PDIM * 2);
  float* stats = (float*)alloc(256 * 4);

  // weight swizzle to WMMA-B layout (once per call, tiny)
  const int wN = DIM * DIM, pN = PDIM * PDIM;
  for (int l = 0; l < NLAYERS; ++l) {
    swizzleB_kernel<<<(wN + 255) / 256, 256, 0, stream>>>(w1 + l * wN,
                                                          w1bf + l * wN, DIM, DIM);
    swizzleB_kernel<<<(wN + 255) / 256, 256, 0, stream>>>(w2 + l * wN,
                                                          w2bf + l * wN, DIM, DIM);
  }
  swizzleB_kernel<<<(pN + 255) / 256, 256, 0, stream>>>(pw1, pw1bf, PDIM, PDIM);
  swizzleB_kernel<<<(pN + 255) / 256, 256, 0, stream>>>(pw2, pw2bf, PDIM, PDIM);
  zero_f32_kernel<<<768, 256, 0, stream>>>(pooled, NGRAPH * PDIM);

  const float* hin = x;
  float* zb[NLAYERS] = {bufA, bufB, bufA};
  for (int l = 0; l < NLAYERS; ++l) {
    float* zl = zb[l];
    zero_f32_kernel<<<4096, 256, 0, stream>>>(agg, N_NODES * DIM);
    scatter_add_kernel<<<(N_EDGES * 32) / 256, 256, 0, stream>>>(hin, srcI,
                                                                 dstI, agg);
    gin_mlp_kernel<<<N_NODES / 16, 256, 0, stream>>>(
        hin, agg, w1bf + l * wN, b1 + l * DIM, w2bf + l * wN, b2 + l * DIM, zl);
    zero_f32_kernel<<<1, 256, 0, stream>>>(stats, 256);
    col_stats_kernel<<<256, 256, 0, stream>>>(zl, stats, stats + DIM, N_NODES);
    bn_apply_pool_kernel<<<(N_NODES * DIM) / 256, 256, 0, stream>>>(
        zl, stats, stats + DIM, gamma + l * DIM, beta + l * DIM, batch, zl,
        pooled, l);
    hin = zl;
  }

  // projection head
  cvt_bf16_kernel<<<(NGRAPH * PDIM + 255) / 256, 256, 0, stream>>>(
      pooled, xcat_bf, NGRAPH * PDIM);
  gemm_bf16_kernel<true, true><<<dim3(NGRAPH / 16, PDIM / 128), 256, 0,
                                 stream>>>(xcat_bf, pw1bf, pb1, nullptr, y1bf,
                                           NGRAPH, PDIM, PDIM);
  gemm_bf16_kernel<false, false><<<dim3(NGRAPH / 16, PDIM / 128), 256, 0,
                                   stream>>>(y1bf, pw2bf, pb2, yfin, nullptr,
                                             NGRAPH, PDIM, PDIM);
  l2norm_kernel<<<2 * NGRAPH, 128, 0, stream>>>(yfin, pooled, out);
}
